// HermiteFuncKANLayer_13271448944840
// MI455X (gfx1250) — compile-verified, matrix operand order checked by reading, not run on hardware
//
#include <hip/hip_runtime.h>
#include <math.h>

typedef __attribute__((ext_vector_type(16))) _Float16 v16h;
typedef __attribute__((ext_vector_type(8)))  _Float16 v8h;
typedef __attribute__((ext_vector_type(8)))  float    v8f;

#define IN_DIM   1024
#define OUT_DIM  1024
#define DEG      8
#define NDEG     9
#define KDIM     (IN_DIM * NDEG)   // 9216
#define IC       32                // input features per K-chunk
#define KC       (IC * NDEG)       // 288 = lcm(32,9) -> 9 k-steps of 32
#define NCHUNK   (IN_DIM / IC)     // 32
#define LDS_STRIDE 296             // 288 + 8 halves pad (stride-4 bank pattern, conflict-free b128)
#define TILE_M   128
#define TILE_N   256
#define MT       4                 // 16-row subtiles per wave
#define NT       4                 // 16-col subtiles per wave
#define CSCALE   1024.0f           // lift coeffs (~1e-4 std) into f16 normal range
#define OSCALE   (1.0f / 1024.0f)

// Hermite recurrence: h_d = C1[d]*t*h_{d-1} - C2[d]*h_{d-2}
constexpr float C1c[NDEG] = {0.f, 0.f, 1.0f, 0.816496580927726f, 0.7071067811865476f,
                             0.6324555320336759f, 0.5773502691896258f,
                             0.5345224838248488f, 0.5f};
constexpr float C2c[NDEG] = {0.f, 0.f, 0.7071067811865476f, 0.816496580927726f,
                             0.8660254037844386f, 0.8944271909999159f,
                             0.9128709291752769f, 0.9258200997725514f,
                             0.9354143466934853f};

// ---- Kernel 1: repack coeffs [i][o][d] f32 -> Bt[o][k=i*9+d] f16 * CSCALE ----
__global__ void repack_coeffs(const float* __restrict__ coeffs,
                              _Float16* __restrict__ bt, int total) {
    int idx = blockIdx.x * blockDim.x + threadIdx.x;
    if (idx >= total) return;
    int o = idx / KDIM;
    int k = idx - o * KDIM;
    int i = k / NDEG;
    int d = k - i * NDEG;
    float v = coeffs[((long)i * OUT_DIM + o) * NDEG + d] * CSCALE;
    bt[idx] = (_Float16)v;
}

// ---- basis for 128 rows x 32 features of chunk c into one LDS buffer ----
__device__ __forceinline__ void basis_chunk(const float* __restrict__ x,
                                            _Float16* __restrict__ buf,
                                            int rowBase, int c, int tid) {
    const int il = tid & 31;          // feature within chunk
    const int r0 = tid >> 5;          // 0..7, step 8 over rows
    const int ig = c * IC + il;
#pragma unroll 2
    for (int p = 0; p < 16; ++p) {
        const int r  = r0 + p * 8;
        float xv = x[(long)(rowBase + r) * IN_DIM + ig];
        // tanh(v) = 1 - 2/(exp(2v)+1): one v_exp_f32 + v_rcp_f32; exact limits at +-inf
        float e2 = __expf(2.0f * xv);
        float th = 1.0f - 2.0f / (e2 + 1.0f);
        float t  = th * 5.123105625617661f;            // * (sqrt(17)+1)
        float g  = __expf(-0.5f * t * t);
        float h0 = 0.7511255444649425f * g;            // pi^-1/4
        float h1 = 1.0622519320271969f * t * g;        // sqrt(2)*pi^-1/4
        _Float16* dst = &buf[r * LDS_STRIDE + il * NDEG];
        dst[0] = (_Float16)h0;
        dst[1] = (_Float16)h1;
        float hm2 = h0, hm1 = h1;
#pragma unroll
        for (int d = 2; d <= DEG; ++d) {
            float h = C1c[d] * t * hm1 - C2c[d] * hm2;
            dst[d] = (_Float16)h;
            hm2 = hm1; hm1 = h;
        }
    }
}

// ---- Kernel 2: fused Hermite-basis + WMMA GEMM, double-buffered LDS ----
__launch_bounds__(256)
__global__ void hermite_kan_gemm(const float* __restrict__ x,
                                 const _Float16* __restrict__ bt,
                                 float* __restrict__ out) {
    __shared__ __align__(16) _Float16 lds_a[2][TILE_M * LDS_STRIDE]; // 2 x 74 KB

    const int tid    = threadIdx.x;
    const int lane   = tid & 31;
    const int wave   = tid >> 5;
    const int wm     = wave & 1;    // 2 row-groups of 64
    const int wn     = wave >> 1;   // 4 col-groups of 64
    const int lane16 = lane & 15;
    const int laneHi = lane >> 4;

    const int rowBase = blockIdx.y * TILE_M;
    const int colBase = blockIdx.x * TILE_N;

    v8f acc[MT][NT];
    const v8f vzero = {0.f, 0.f, 0.f, 0.f, 0.f, 0.f, 0.f, 0.f};
#pragma unroll
    for (int mt = 0; mt < MT; ++mt)
#pragma unroll
        for (int nt = 0; nt < NT; ++nt) acc[mt][nt] = vzero;

    basis_chunk(x, lds_a[0], rowBase, 0, tid);
    __syncthreads();

    for (int c = 0; c < NCHUNK; ++c) {
        // Prefill next chunk's basis into the other buffer; its VALU/TRANS work
        // co-executes with this chunk's WMMAs across waves (no barrier until end).
        if (c + 1 < NCHUNK)
            basis_chunk(x, lds_a[(c + 1) & 1], rowBase, c + 1, tid);

        const _Float16* abuf = lds_a[c & 1];
        for (int ks = 0; ks < NDEG; ++ks) {
            const int klo = ks * 32 + (laneHi ? 8 : 0);

            v16h afrag[MT];
#pragma unroll
            for (int mt = 0; mt < MT; ++mt) {
                const _Float16* src = &abuf[(wm * 64 + mt * 16 + lane16) * LDS_STRIDE + klo];
                union { v16h v; v8h h[2]; } u;
                u.h[0] = *(const v8h*)(src);        // ds_load_b128
                u.h[1] = *(const v8h*)(src + 16);
                afrag[mt] = u.v;
            }

            v16h bfrag[NT];
            const long kg = (long)c * KC + klo;
#pragma unroll
            for (int nt = 0; nt < NT; ++nt) {
                const int col = colBase + wn * 64 + nt * 16 + lane16;
                const _Float16* src = &bt[(long)col * KDIM + kg];
                union { v16h v; v8h h[2]; } u;
                u.h[0] = *(const v8h*)(src);        // global_load_b128 (L2-resident)
                u.h[1] = *(const v8h*)(src + 16);
                bfrag[nt] = u.v;
            }

#pragma unroll
            for (int mt = 0; mt < MT; ++mt)
#pragma unroll
                for (int nt = 0; nt < NT; ++nt)
                    acc[mt][nt] = __builtin_amdgcn_wmma_f32_16x16x32_f16(
                        false, afrag[mt], false, bfrag[nt],
                        (short)0, acc[mt][nt], false, false);
        }
        __syncthreads();   // buf[(c+1)&1] writes visible; buf[c&1] reads done
    }

    // ---- Epilogue: C/D layout -> row = base + r + 8*laneHi, col = lane16 ----
#pragma unroll
    for (int mt = 0; mt < MT; ++mt) {
#pragma unroll
        for (int nt = 0; nt < NT; ++nt) {
            const int col  = colBase + wn * 64 + nt * 16 + lane16;
            const int row0 = rowBase + wm * 64 + mt * 16 + laneHi * 8;
#pragma unroll
            for (int r = 0; r < 8; ++r)
                out[(long)(row0 + r) * OUT_DIM + col] = acc[mt][nt][r] * OSCALE;
        }
    }
}

extern "C" void kernel_launch(void* const* d_in, const int* in_sizes, int n_in,
                              void* d_out, int out_size, void* d_ws, size_t ws_size,
                              hipStream_t stream) {
    const float* x      = (const float*)d_in[0];
    const float* coeffs = (const float*)d_in[1];
    float*       out    = (float*)d_out;
    _Float16*    bt     = (_Float16*)d_ws;     // needs OUT_DIM*KDIM*2 = 18.9 MB

    const int rows  = in_sizes[0] / IN_DIM;    // 8192
    const int total = OUT_DIM * KDIM;

    repack_coeffs<<<(total + 255) / 256, 256, 0, stream>>>(coeffs, bt, total);

    dim3 grid(OUT_DIM / TILE_N, rows / TILE_M);
    hermite_kan_gemm<<<grid, dim3(256), 0, stream>>>(x, bt, out);
}
